// QuantLinear_83631603188256
// MI455X (gfx1250) — matrix operand order repaired and checked by reading, not compiled, not run on hardware
//
#include <hip/hip_runtime.h>

// ---------------------------------------------------------------------------
// QuantLinear with butterfly transforms, MI455X (gfx1250, wave32, WMMA)
//   d_in: x[16,8192]f16, qweight[1024,8192]i32, scale[1,8192]f16,
//         scale_hw[8192]f16, v1[64,64]f16, v2[128,128]f16, u1[64,64]f16,
//         u2[128,128]f16, bias[8192]f16, v_inp, v_outp, u_inp, u_outp (i32)
//   d_out: [16,8192] float
// ---------------------------------------------------------------------------

#define IN_F   8192
#define OUT_F  8192
#define BATCH  16
#define KSPLIT 8
#define KCHUNK (IN_F / KSPLIT)   // 1024

typedef __attribute__((ext_vector_type(16))) _Float16 v16h;
typedef __attribute__((ext_vector_type(8)))  _Float16 v8h;
typedef __attribute__((ext_vector_type(8)))  float    v8f;

// ---------------------------------------------------------------------------
// Kernel A: xs = permute_out( v1 @ (x/scale_hw)[v_inp].reshape(64,128) @ v2^T )
// One workgroup per batch row. LDS: 16KB half + 32KB float = 48KB.
// ---------------------------------------------------------------------------
__global__ __launch_bounds__(256)
void input_butterfly_kernel(const _Float16* __restrict__ x,
                            const _Float16* __restrict__ scale_hw,
                            const _Float16* __restrict__ v1,
                            const _Float16* __restrict__ v2,
                            const int* __restrict__ v_inp,
                            const int* __restrict__ v_outp,
                            _Float16* __restrict__ xs)
{
    __shared__ _Float16 gh[IN_F];   // 16 KB
    __shared__ float    tf[IN_F];   // 32 KB
    const int b   = blockIdx.x;
    const int tid = threadIdx.x;

    // gather + scale_hw divide
    for (int j = tid; j < IN_F; j += 256) {
        const int src = v_inp[j];
        gh[j] = (_Float16)((float)x[b * IN_F + src] / (float)scale_hw[src]);
    }
    __syncthreads();

    // t1[i,k] = sum_j v1[i,j] * g[j,k]      (64x64)(64x128)
    for (int idx = tid; idx < IN_F; idx += 256) {
        const int i = idx >> 7, k = idx & 127;
        float acc = 0.f;
        #pragma unroll 8
        for (int j = 0; j < 64; ++j)
            acc += (float)v1[i * 64 + j] * (float)gh[j * 128 + k];
        tf[idx] = acc;
    }
    __syncthreads();

    // t2[i,l] = sum_k t1[i,k] * v2[l,k]     (64x128)(128x128)^T
    for (int idx = tid; idx < IN_F; idx += 256) {
        const int i = idx >> 7, l = idx & 127;
        float acc = 0.f;
        #pragma unroll 8
        for (int k = 0; k < 128; ++k)
            acc += tf[i * 128 + k] * (float)v2[l * 128 + k];
        gh[idx] = (_Float16)acc;     // gh dead after phase 2; reuse
    }
    __syncthreads();

    // scatter via v_outp
    for (int m = tid; m < IN_F; m += 256)
        xs[b * IN_F + m] = gh[v_outp[m]];
}

// ---------------------------------------------------------------------------
// Kernel B: split-K dequant GEMM.  Y_part[kc] = xs[:,Krange] @ W[Krange,:]
// grid.x = OUT_F/128 (8 waves/block, 16 cols/wave), grid.y = KSPLIT.
// Streams qweight once, dequants nibbles in-lane, v_wmma_f32_16x16x32_f16.
// K-loop unrolled x4 so ~16 loads are in flight per WMMA group.
// ---------------------------------------------------------------------------
__global__ __launch_bounds__(256)
void dequant_wmma_gemm_kernel(const _Float16* __restrict__ xs,
                              const int* __restrict__ qw,
                              const _Float16* __restrict__ scale,
                              float* __restrict__ ypart)
{
    const int lane  = threadIdx.x & 31;
    const int wid   = threadIdx.x >> 5;
    const int ntile = blockIdx.x * 128 + wid * 16;
    const int kc    = blockIdx.y;                 // K chunk id
    const int n     = ntile + (lane & 15);

    // A-operand addressing (16-bit A 16x32 layout, wave32)
    const int m     = lane & 15;
    const int khalf = (lane >> 4) * 8;            // K sub-offset for A halves
    const int koff  = (lane >> 4) * 8;            // K sub-offset for B column

    // dequant constants for this lane's column
    const float s  = (float)scale[n];
    const float da = s * (2.0f / 15.0f);
    const float db = -s;

    const _Float16* __restrict__ xrow = xs + m * IN_F + khalf;
    const int*      __restrict__ qcol = qw + n;   // row stride = OUT_F int32s

    const int kbeg = kc * KCHUNK;
    const int kend = kbeg + KCHUNK;

    v8f acc = {};
    #pragma unroll 4
    for (int kt = kbeg; kt < kend; kt += 32) {
        // ---- A tile: lane holds K=[kt+khalf,+8) and K=[kt+16+khalf,+8)
        v8h alo = *(const v8h*)(xrow + kt);
        v8h ahi = *(const v8h*)(xrow + kt + 16);
        v16h a  = __builtin_shufflevector(alo, ahi,
                                          0, 1, 2, 3, 4, 5, 6, 7,
                                          8, 9, 10, 11, 12, 13, 14, 15);

        // ---- B tile: lane holds column n, K=[kt+koff,+8) and K=[kt+16+koff,+8)
        //      = two packed int32 rows of qweight (8 nibbles each)
        const int r0 = (kt + koff) >> 3;          // qweight row (8 codes/row)
        const int q0 = qcol[(size_t)r0 * OUT_F];
        const int q1 = qcol[(size_t)(r0 + 2) * OUT_F];
        __builtin_prefetch(qcol + (size_t)(r0 + 64) * OUT_F, 0, 0);

        v16h bq;
        #pragma unroll
        for (int e = 0; e < 8; ++e) {
            bq[e]     = (_Float16)fmaf((float)((q0 >> (4 * e)) & 15), da, db);
            bq[e + 8] = (_Float16)fmaf((float)((q1 >> (4 * e)) & 15), da, db);
        }

        // D = A x B + C   (16x16x32 f16 -> f32)
        acc = __builtin_amdgcn_wmma_f32_16x16x32_f16(
                  false, a, false, bq, (short)0, acc, false, false);
    }

    // ---- store C/D tile: VGPR r -> M = r + (lane>=16)*8, N = lane%16
    float* __restrict__ yp = ypart + (size_t)kc * (BATCH * OUT_F);
    const int mbase = (lane >> 4) * 8;
    #pragma unroll
    for (int r = 0; r < 8; ++r)
        yp[(size_t)(mbase + r) * OUT_F + n] = acc[r];
}

// ---------------------------------------------------------------------------
// Kernel B2: deterministic fixed-order reduction of KSPLIT partials -> f16 Y
// ---------------------------------------------------------------------------
__global__ __launch_bounds__(256)
void splitk_reduce_kernel(const float* __restrict__ ypart,
                          _Float16* __restrict__ y)
{
    const int idx = blockIdx.x * 256 + threadIdx.x;   // 0 .. BATCH*OUT_F-1
    float acc = ypart[idx];
    #pragma unroll
    for (int c = 1; c < KSPLIT; ++c)
        acc += ypart[(size_t)c * (BATCH * OUT_F) + idx];
    y[idx] = (_Float16)acc;
}

// ---------------------------------------------------------------------------
// Kernel C: out = ( u1^T @ Y[u_outp].reshape(64,128) @ u2 )[u_inp] + bias
// ---------------------------------------------------------------------------
__global__ __launch_bounds__(256)
void output_butterfly_kernel(const _Float16* __restrict__ y,
                             const _Float16* __restrict__ u1,
                             const _Float16* __restrict__ u2,
                             const _Float16* __restrict__ bias,
                             const int* __restrict__ u_inp,
                             const int* __restrict__ u_outp,
                             float* __restrict__ out)
{
    __shared__ _Float16 gh[OUT_F];  // 16 KB
    __shared__ float    tf[OUT_F];  // 32 KB
    const int b   = blockIdx.x;
    const int tid = threadIdx.x;

    for (int j = tid; j < OUT_F; j += 256)
        gh[j] = y[b * OUT_F + u_outp[j]];
    __syncthreads();

    // t1[i,k] = sum_j u1[j,i] * g[j,k]   (u1 transposed)
    for (int idx = tid; idx < OUT_F; idx += 256) {
        const int i = idx >> 7, k = idx & 127;
        float acc = 0.f;
        #pragma unroll 8
        for (int j = 0; j < 64; ++j)
            acc += (float)u1[j * 64 + i] * (float)gh[j * 128 + k];
        tf[idx] = acc;
    }
    __syncthreads();

    // t2[i,l] = sum_k t1[i,k] * u2[k,l]
    for (int idx = tid; idx < OUT_F; idx += 256) {
        const int i = idx >> 7, l = idx & 127;
        float acc = 0.f;
        #pragma unroll 8
        for (int k = 0; k < 128; ++k)
            acc += tf[i * 128 + k] * (float)u2[k * 128 + l];
        gh[idx] = (_Float16)acc;
    }
    __syncthreads();

    // final gather + bias (fp16 add, like reference), emit as f32
    for (int mo = tid; mo < OUT_F; mo += 256) {
        _Float16 v = (_Float16)(gh[u_inp[mo]] + bias[mo]);
        out[b * OUT_F + mo] = (float)v;
    }
}

// ---------------------------------------------------------------------------
extern "C" void kernel_launch(void* const* d_in, const int* in_sizes, int n_in,
                              void* d_out, int out_size, void* d_ws, size_t ws_size,
                              hipStream_t stream)
{
    const _Float16* x        = (const _Float16*)d_in[0];
    const int*      qweight  = (const int*)     d_in[1];
    const _Float16* scale    = (const _Float16*)d_in[2];
    const _Float16* scale_hw = (const _Float16*)d_in[3];
    const _Float16* v1       = (const _Float16*)d_in[4];
    const _Float16* v2       = (const _Float16*)d_in[5];
    const _Float16* u1       = (const _Float16*)d_in[6];
    const _Float16* u2       = (const _Float16*)d_in[7];
    const _Float16* bias     = (const _Float16*)d_in[8];
    const int*      v_inp    = (const int*)d_in[9];
    const int*      v_outp   = (const int*)d_in[10];
    const int*      u_inp    = (const int*)d_in[11];
    const int*      u_outp   = (const int*)d_in[12];

    // workspace layout
    char* ws = (char*)d_ws;
    _Float16* xs    = (_Float16*)(ws);                              // 256 KB
    _Float16* ybuf  = (_Float16*)(ws + (size_t)BATCH * IN_F * 2);   // 256 KB
    float*    ypart = (float*)   (ws + (size_t)BATCH * IN_F * 4);   // 4 MB

    // (1) input butterfly -> xs
    input_butterfly_kernel<<<BATCH, 256, 0, stream>>>(
        x, scale_hw, v1, v2, v_inp, v_outp, xs);

    // (2) split-K dequant + WMMA GEMM -> fp32 partials
    dim3 grid_b(OUT_F / 128, KSPLIT);
    dequant_wmma_gemm_kernel<<<grid_b, 256, 0, stream>>>(
        xs, qweight, scale, ypart);

    // (2b) deterministic reduce -> fp16 Y
    splitk_reduce_kernel<<<(BATCH * OUT_F) / 256, 256, 0, stream>>>(
        ypart, ybuf);

    // (3) output butterfly + bias -> d_out (f32)
    output_butterfly_kernel<<<BATCH, 256, 0, stream>>>(
        ybuf, u1, u2, bias, u_inp, u_outp, (float*)d_out);
}
